// STA_GC_910533067542
// MI455X (gfx1250) — matrix operand order repaired and checked by reading, not compiled, not run on hardware
//
#include <hip/hip_runtime.h>
#include <hip/hip_bf16.h>

// ===========================================================================
// MI455X (gfx1250, wave32) implementation of the STA-GC reference.
// Heavy matmuls use v_wmma_f32_16x16x32_bf16 with LDS tiles staged directly
// in WMMA fragment layout (per-lane contiguous -> ds_load_b128 fragment
// fetches, packed b32 staging stores).  Small V=25 graph contractions and
// 5-tap temporal convs are VALU kernels.
// ===========================================================================

typedef __attribute__((ext_vector_type(16))) __bf16 v16bf;
typedef __attribute__((ext_vector_type(8)))  float  v8f;

__device__ __forceinline__ unsigned short f2bf(float f) {
  unsigned u = __float_as_uint(f);
  u = u + 0x7FFFu + ((u >> 16) & 1u);          // round-to-nearest-even
  return (unsigned short)(u >> 16);
}
__device__ __forceinline__ unsigned pack2bf(float f0, float f1) {
  return (unsigned)f2bf(f0) | ((unsigned)f2bf(f1) << 16);
}

// ---------------------------------------------------------------------------
// Generic WMMA GEMM:  O[m,p] = epilogue( sum_k A[m,k] * B[k,p] )
// A element addr = A + z*zStrideA + m*sAm + k*sAk
// B element addr = B + z*zStrideB + k*sBk + (p/colDiv)*colStrideT + (p%colDiv)*colStrideV
// Optional: fold relu(B*inScale[k]+inShift[k]) on load; bias[m]; out-BN;
//           residual add; relu.  Out/Res use the same column-mapping family.
// ---------------------------------------------------------------------------
struct GemmArgs {
  const float* A; long zStrideA; long sAm; long sAk;
  const float* B; long zStrideB; long sBk;
  int  colDiv; long colStrideT; long colStrideV;
  const float* bias;
  const float* inScale; const float* inShift; int inFold;
  const float* bnScale; const float* bnShift; int bnOut;
  const float* R; long zStrideR; long sRm; int rDiv; long rStrideT; long rStrideV;
  float* O; long zStrideO; long sOm; int oDiv; long oStrideT; long oStrideV;
  int M, K, P, relu;
};

#define BM 64
#define BN 64
#define BK 32
#define FRAG_PITCH 24   // ushorts per lane row (16 used + pad); 48B keeps 16B align

__device__ __forceinline__ void gemm_store(const GemmArgs& g, float* O,
                                           const float* R, int m, int p, float v) {
  if (p >= g.P) return;
  if (g.bias)  v += g.bias[m];
  if (g.bnOut) v = v * g.bnScale[m] + g.bnShift[m];
  if (R) {
    long ro = (long)m * g.sRm + (long)(p / g.rDiv) * g.rStrideT
            + (long)(p % g.rDiv) * g.rStrideV;
    v += R[ro];
  }
  if (g.relu) v = v > 0.f ? v : 0.f;
  long oo = (long)m * g.sOm + (long)(p / g.oDiv) * g.oStrideT
          + (long)(p % g.oDiv) * g.oStrideV;
  O[oo] = v;
}

__global__ __launch_bounds__(256) void gemm_bf16_wmma_kernel(GemmArgs g) {
  // LDS tiles in WMMA *fragment* layout: [16x16-tile][lane][slot]
  __shared__ unsigned short Asf[4][32][FRAG_PITCH];
  __shared__ unsigned short Bsf[4][32][FRAG_PITCH];
  const int tid  = threadIdx.x;
  const int wave = tid >> 5, lane = tid & 31;
  const int m0 = blockIdx.y * BM;
  const int p0 = blockIdx.x * BN;
  const float* A = g.A + (long)blockIdx.z * g.zStrideA;
  const float* B = g.B + (long)blockIdx.z * g.zStrideB;
  float*       O = g.O + (long)blockIdx.z * g.zStrideO;
  const float* R = g.R ? (g.R + (long)blockIdx.z * g.zStrideR) : nullptr;

  const int wm = wave & 3;   // 4 m-subtiles of 16
  const int wn = wave >> 2;  // 2 column groups of 32 (= 2 n-subtiles each)
  const bool fullM = (m0 + BM) <= g.M;
  const bool fullP = (p0 + BN) <= g.P;
  v8f acc0 = {}; v8f acc1 = {};

  for (int k0 = 0; k0 < g.K; k0 += BK) {
    const bool fullK = (k0 + BK) <= g.K;
    // ---- stage A tile (f32 -> bf16) directly into fragment layout ----
    // element (mm,kk): lane = ((kk>>3)&1)*16 + (mm&15); slot = ((kk>>4)<<3)|(kk&7)
    if (fullM & fullK) {
      for (int it = tid; it < (BM * BK) / 2; it += 256) {
        int mm = it >> 4, kk = (it & 15) << 1;
        const float* p = A + (long)(m0 + mm) * g.sAm + (long)(k0 + kk) * g.sAk;
        unsigned pk = pack2bf(p[0], p[g.sAk]);
        *(unsigned*)&Asf[mm >> 4][(((kk >> 3) & 1) << 4) + (mm & 15)]
                       [((kk >> 4) << 3) | (kk & 7)] = pk;
      }
    } else {
      for (int it = tid; it < (BM * BK) / 2; it += 256) {
        int mm = it >> 4, kk = (it & 15) << 1;
        int mg = m0 + mm, kg = k0 + kk;
        float f0 = 0.f, f1 = 0.f;
        if (mg < g.M) {
          if (kg < g.K)     f0 = A[(long)mg * g.sAm + (long)kg * g.sAk];
          if (kg + 1 < g.K) f1 = A[(long)mg * g.sAm + (long)(kg + 1) * g.sAk];
        }
        *(unsigned*)&Asf[mm >> 4][(((kk >> 3) & 1) << 4) + (mm & 15)]
                       [((kk >> 4) << 3) | (kk & 7)] = pack2bf(f0, f1);
      }
    }
    // ---- stage B tile (f32 -> bf16) into fragment layout ----
    // element (kk,pp): lane = ((kk>>3)&1)*16 + (pp&15); slot = ((kk>>4)<<3)|(kk&7)
    if (fullP & fullK & !g.inFold) {
      for (int it = tid; it < (BK * BN) / 2; it += 256) {
        int pp = it & 63, kk = (it >> 6) << 1;
        int pg = p0 + pp;
        long off = (long)(pg / g.colDiv) * g.colStrideT
                 + (long)(pg % g.colDiv) * g.colStrideV;
        const float* p = B + (long)(k0 + kk) * g.sBk + off;
        unsigned pk = pack2bf(p[0], p[g.sBk]);
        *(unsigned*)&Bsf[pp >> 4][(((kk >> 3) & 1) << 4) + (pp & 15)][((kk >> 4) << 3) | (kk & 7)] = pk;
      }
    } else {
      for (int it = tid; it < (BK * BN) / 2; it += 256) {
        int pp = it & 63, kk = (it >> 6) << 1;
        int kg = k0 + kk, pg = p0 + pp;
        float f0 = 0.f, f1 = 0.f;
        if (pg < g.P) {
          long off = (long)(pg / g.colDiv) * g.colStrideT
                   + (long)(pg % g.colDiv) * g.colStrideV;
          if (kg < g.K) {
            f0 = B[(long)kg * g.sBk + off];
            if (g.inFold) f0 = fmaxf(f0 * g.inScale[kg] + g.inShift[kg], 0.f);
          }
          if (kg + 1 < g.K) {
            f1 = B[(long)(kg + 1) * g.sBk + off];
            if (g.inFold) f1 = fmaxf(f1 * g.inScale[kg + 1] + g.inShift[kg + 1], 0.f);
          }
        }
        *(unsigned*)&Bsf[pp >> 4][(((kk >> 3) & 1) << 4) + (pp & 15)][((kk >> 4) << 3) | (kk & 7)] = pack2bf(f0, f1);
      }
    }
    // prefetch next k-tile of B (contiguous-column cases)
    if (g.colStrideV == 1 && tid < 64) {
      int kn = k0 + BK + (tid >> 1);
      if (kn < g.K) {
        int pg = p0 + ((tid & 1) << 5);
        long off = (long)(pg / g.colDiv) * g.colStrideT + (long)(pg % g.colDiv);
        __builtin_prefetch(B + (long)kn * g.sBk + off, 0, 0);
      }
    }
    __syncthreads();

    // ---- per-lane fragment fetch: contiguous 32B -> 2x ds_load_b128 ----
    union Frag { v16bf v; uint4 q[2]; } a, b0, b1;
    {
      const unsigned short* ap = &Asf[wm][lane][0];
      a.q[0] = *(const uint4*)ap;  a.q[1] = *(const uint4*)(ap + 8);
      const unsigned short* bp0 = &Bsf[wn * 2][lane][0];
      b0.q[0] = *(const uint4*)bp0; b0.q[1] = *(const uint4*)(bp0 + 8);
      const unsigned short* bp1 = &Bsf[wn * 2 + 1][lane][0];
      b1.q[0] = *(const uint4*)bp1; b1.q[1] = *(const uint4*)(bp1 + 8);
    }
    acc0 = __builtin_amdgcn_wmma_f32_16x16x32_bf16(false, a.v, false, b0.v,
                                                   (short)0, acc0, false, false);
    acc1 = __builtin_amdgcn_wmma_f32_16x16x32_bf16(false, a.v, false, b1.v,
                                                   (short)0, acc1, false, false);
    __syncthreads();
  }

  // C/D layout: lane owns rows mBase..mBase+7, col = lane&15 (+16 for acc1 tile)
  int colA  = p0 + wn * 32 + (lane & 15);
  int colB  = colA + 16;
  int mBase = m0 + wm * 16 + (lane >> 4) * 8;
#pragma unroll
  for (int r = 0; r < 8; ++r) {
    int m = mBase + r;
    if (m < g.M) {
      gemm_store(g, O, R, m, colA, acc0[r]);
      gemm_store(g, O, R, m, colB, acc1[r]);
    }
  }
}

// ---------------------------------------------------------------------------
// BN param prep: scale = g*rsqrt(v+eps); shift = b - m*scale
// ---------------------------------------------------------------------------
__global__ void bn_prep_kernel(const float* g_, const float* b_, const float* m_,
                               const float* v_, float* scale, float* shift, int n) {
  int i = blockIdx.x * blockDim.x + threadIdx.x;
  if (i < n) {
    float s = g_[i] * rsqrtf(v_[i] + 1e-5f);
    scale[i] = s;
    shift[i] = b_[i] - m_[i] * s;
  }
}

// ---------------------------------------------------------------------------
// SGC adjacency contraction: z[c,t,w] = sum_{k,v} y[k*128+c,t,v]*A[k,v,w]*E[k,v,w]
// then relu(bn(z) + res).  Per-n launch (yb is one n slice).
// ---------------------------------------------------------------------------
__global__ void sgc_adj_kernel(const float* yb, const float* A, const float* E,
                               const float* scale, const float* shift,
                               const float* res, float* out) {
  __shared__ float Ae[1250];
  for (int i = threadIdx.x; i < 1250; i += blockDim.x) Ae[i] = A[i] * E[i];
  __syncthreads();
  int idx = blockIdx.x * blockDim.x + threadIdx.x;
  if (idx >= 128 * 288 * 25) return;
  int w = idx % 25, t = (idx / 25) % 288, c = idx / (25 * 288);
  float acc = 0.f;
#pragma unroll
  for (int k = 0; k < 2; ++k) {
    const float* yr = yb + ((long)(k * 128 + c) * 288 + t) * 25;
    const float* ae = Ae + k * 625 + w;
#pragma unroll
    for (int v = 0; v < 25; ++v) acc += yr[v] * ae[v * 25];
  }
  float r = acc * scale[c] + shift[c] + res[idx];
  out[idx] = r > 0.f ? r : 0.f;
}

// ---------------------------------------------------------------------------
// SAGC attention: dA[n,t,v,w] = l2norm_w( sum_q ph[n,q,t,v]*ps[n,q,t,w] )
// one wave per (t,n)
// ---------------------------------------------------------------------------
__global__ void sagc_attn_kernel(const float* ph, const float* ps, float* dA) {
  __shared__ float P1[800], P2[800];
  int t = blockIdx.x, n = blockIdx.y;
  const float* p1 = ph + (long)n * 32 * 7200 + t * 25;
  const float* p2 = ps + (long)n * 32 * 7200 + t * 25;
  for (int i = threadIdx.x; i < 800; i += 32) {
    int q = i / 25, v = i % 25;
    P1[i] = p1[(long)q * 7200 + v];
    P2[i] = p2[(long)q * 7200 + v];
  }
  __syncthreads();
  int v = threadIdx.x;
  if (v < 25) {
    float row[25]; float ss = 0.f;
#pragma unroll
    for (int w = 0; w < 25; ++w) {
      float a = 0.f;
#pragma unroll
      for (int q = 0; q < 32; ++q) a += P1[q * 25 + v] * P2[q * 25 + w];
      row[w] = a; ss += a * a;
    }
    float inv = 1.f / fmaxf(sqrtf(ss), 1e-12f);
    float* o = dA + (((long)n * 288 + t) * 25 + v) * 25;
#pragma unroll
    for (int w = 0; w < 25; ++w) o[w] = row[w] * inv;
  }
}

// SAGC apply: out = relu(bn(sum_v y[c,t,v]*dA[t,v,w]) + x1)
__global__ void sagc_apply_kernel(const float* y, const float* dA,
                                  const float* scale, const float* shift,
                                  const float* x1, float* out) {
  long n = blockIdx.y;
  int idx = blockIdx.x * blockDim.x + threadIdx.x;
  if (idx >= 128 * 288 * 25) return;
  int w = idx % 25, t = (idx / 25) % 288, c = idx / (25 * 288);
  const float* yr = y + n * 128 * 7200 + ((long)c * 288 + t) * 25;
  const float* ar = dA + ((n * 288 + t) * 25) * 25 + w;
  float acc = 0.f;
#pragma unroll
  for (int v = 0; v < 25; ++v) acc += yr[v] * ar[v * 25];
  long o = n * 128 * 7200 + idx;
  float r = acc * scale[c] + shift[c] + x1[o];
  out[o] = r > 0.f ? r : 0.f;
}

// l2-normalize rows (one wave per row)
__global__ void l2norm_rows_kernel(float* dA, int rows, int len) {
  int row  = blockIdx.x * (blockDim.x >> 5) + (threadIdx.x >> 5);
  int lane = threadIdx.x & 31;
  if (row >= rows) return;
  float* r = dA + (long)row * len;
  float ss = 0.f;
  for (int i = lane; i < len; i += 32) { float v = r[i]; ss += v * v; }
  for (int off = 16; off; off >>= 1) ss += __shfl_xor(ss, off);
  float inv = 1.f / fmaxf(sqrtf(ss), 1e-12f);
  for (int i = lane; i < len; i += 32) r[i] *= inv;
}

// 5-tap dilated temporal conv (32->32 ch) + bias + BN, writes concat slice
__global__ void tconv_kernel(const float* h, const float* w2, const float* b2,
                             const float* scale, const float* shift, float* out,
                             int Tin, int Tout, int stride, int dil, int chOff) {
  __shared__ float W[32 * 32 * 5];
  for (int i = threadIdx.x; i < 5120; i += blockDim.x) W[i] = w2[i];
  __syncthreads();
  int idx = blockIdx.x * blockDim.x + threadIdx.x;
  if (idx >= 32 * Tout * 25) return;
  int v = idx % 25, tp = (idx / 25) % Tout, o = idx / (25 * Tout);
  const float* hn = h + (long)blockIdx.y * 32 * Tin * 25;
  float acc = b2[o];
  int tbase = tp * stride - 2 * dil;
#pragma unroll
  for (int kh = 0; kh < 5; ++kh) {
    int t = tbase + kh * dil;
    if (t < 0 || t >= Tin) continue;
    const float* hc = hn + t * 25 + v;
    const float* wr = W + o * 160 + kh;
    for (int i2 = 0; i2 < 32; ++i2) acc += hc[(long)i2 * Tin * 25] * wr[i2 * 5];
  }
  float r = acc * scale[o] + shift[o];
  out[(long)blockIdx.y * 192 * Tout * 25 + (long)(chOff + o) * Tout * 25 + tp * 25 + v] = r;
}

// temporal maxpool (win 5, pad 2) + BN, writes concat slice
__global__ void maxpool_kernel(const float* h, const float* scale, const float* shift,
                               float* out, int Tin, int Tout, int stride, int chOff) {
  int idx = blockIdx.x * blockDim.x + threadIdx.x;
  if (idx >= 32 * Tout * 25) return;
  int v = idx % 25, tp = (idx / 25) % Tout, o = idx / (25 * Tout);
  const float* hc = h + (long)blockIdx.y * 32 * Tin * 25 + (long)o * Tin * 25;
  float m = -3.4e38f;
  int tbase = tp * stride - 2;
#pragma unroll
  for (int j = 0; j < 5; ++j) {
    int t = tbase + j;
    if (t >= 0 && t < Tin) { float x = hc[t * 25 + v]; m = x > m ? x : m; }
  }
  float r = m * scale[o] + shift[o];
  out[(long)blockIdx.y * 192 * Tout * 25 + (long)(chOff + o) * Tout * 25 + tp * 25 + v] = r;
}

// ===========================================================================
extern "C" void kernel_launch(void* const* d_in, const int* in_sizes, int n_in,
                              void* d_out, int out_size, void* d_ws, size_t ws_size,
                              hipStream_t stream) {
  if (n_in < 185) return;
  const float* IN[185];
  for (int i = 0; i < 185; ++i) IN[i] = (const float*)d_in[i];
  const int BIG = 1 << 30;
  const long TV = 7200;  // 288*25

  float* W = (float*)d_ws;
  size_t off = 0;
  auto alloc = [&](size_t n) { float* p = W + off; off += n; return p; };
  float* bufA  = alloc(29491200);  // [32][128][7200]
  float* bufB  = alloc(29491200);
  float* bufR  = alloc(29491200);
  float* bufY  = alloc(29491200);  // also aliased as per-n [256][7200] during SGC
  float* bufP1 = alloc(7372800);   // [32][32][7200]
  float* bufP2 = alloc(7372800);
  float* bufDA = alloc(5760000);   // max(sagc full, tagc per-n 25*288*288)
  float* bufC  = alloc(22118400);  // [32][192][3600]
  float* bufH  = alloc(7372800);   // [32][32][7200]
  float* bnb   = alloc(31 * 512);  // 31 BN slots: scale[256] + shift[256]

  auto bscale = [&](int s) { return bnb + (long)s * 512; };
  auto bshift = [&](int s) { return bnb + (long)s * 512 + 256; };
  auto bnprep = [&](int idx, int slot, int n) {
    hipLaunchKernelGGL(bn_prep_kernel, dim3((n + 63) / 64), dim3(64), 0, stream,
                       IN[idx], IN[idx + 1], IN[idx + 2], IN[idx + 3],
                       bscale(slot), bshift(slot), n);
  };
  auto launch_gemm = [&](const GemmArgs& g, int z) {
    dim3 grid((g.P + BN - 1) / BN, (g.M + BM - 1) / BM, z);
    hipLaunchKernelGGL(gemm_bf16_wmma_kernel, grid, dim3(256), 0, stream, g);
  };

  // ---- BN precompute (31 sets) ----
  bnprep(5, 0, 128); bnprep(11, 1, 128); bnprep(19, 2, 128); bnprep(27, 3, 128);
  for (int blk = 0; blk < 2; ++blk) {
    int pb = blk ? 111 : 31, s0 = blk ? 18 : 4;
    for (int br = 0; br < 4; ++br) {
      bnprep(pb + br * 12 + 2, s0 + 2 * br, 32);
      bnprep(pb + br * 12 + 8, s0 + 2 * br + 1, 32);
    }
    bnprep(pb + 50, s0 + 8, 32);  bnprep(pb + 54, s0 + 9, 32);   // b4 bn1/bn2
    bnprep(pb + 60, s0 + 10, 32);                                 // b5 bn1
    bnprep(pb + 64, s0 + 11, 192); bnprep(pb + 70, s0 + 12, 128); // t_bn_in/out
    if (!blk) bnprep(pb + 76, 17, 128);                           // mstl1 res_bn
  }

  // ================= SGC =================
  { // residual: BN(conv1x1(x, res_w, res_b))
    GemmArgs g{};
    g.A = IN[9]; g.sAm = 64; g.sAk = 1;
    g.B = IN[0]; g.sBk = TV; g.zStrideB = 64 * TV;
    g.colDiv = BIG; g.colStrideV = 1;
    g.bias = IN[10];
    g.bnScale = bscale(1); g.bnShift = bshift(1); g.bnOut = 1;
    g.O = bufR; g.sOm = TV; g.zStrideO = 128 * TV; g.oDiv = BIG; g.oStrideV = 1;
    g.M = 128; g.K = 64; g.P = 7200;
    launch_gemm(g, 32);
  }
  for (int n = 0; n < 32; ++n) {  // per-n: 256-ch gcn GEMM, then adjacency fuse
    GemmArgs g{};
    g.A = IN[3]; g.sAm = 64; g.sAk = 1;
    g.B = IN[0] + (long)n * 64 * TV; g.sBk = TV;
    g.colDiv = BIG; g.colStrideV = 1;
    g.bias = IN[4];
    g.O = bufY; g.sOm = TV; g.oDiv = BIG; g.oStrideV = 1;
    g.M = 256; g.K = 64; g.P = 7200;
    launch_gemm(g, 1);
    hipLaunchKernelGGL(sgc_adj_kernel, dim3(3600), dim3(256), 0, stream,
                       bufY, IN[1], IN[2], bscale(0), bshift(0),
                       bufR + (long)n * 128 * TV, bufA + (long)n * 128 * TV);
  }

  // ================= SAGC =================
  for (int ph = 0; ph < 2; ++ph) {  // phi -> bufP1, psi -> bufP2
    GemmArgs g{};
    g.A = IN[15 + ph]; g.sAm = 128; g.sAk = 1;
    g.B = bufA; g.sBk = TV; g.zStrideB = 128 * TV;
    g.colDiv = BIG; g.colStrideV = 1;
    g.O = ph ? bufP2 : bufP1; g.sOm = TV; g.zStrideO = 32 * TV;
    g.oDiv = BIG; g.oStrideV = 1;
    g.M = 32; g.K = 128; g.P = 7200;
    launch_gemm(g, 32);
  }
  hipLaunchKernelGGL(sagc_attn_kernel, dim3(288, 32), dim3(32), 0, stream,
                     bufP1, bufP2, bufDA);
  { // gcn conv
    GemmArgs g{};
    g.A = IN[17]; g.sAm = 128; g.sAk = 1;
    g.B = bufA; g.sBk = TV; g.zStrideB = 128 * TV;
    g.colDiv = BIG; g.colStrideV = 1;
    g.bias = IN[18];
    g.O = bufY; g.sOm = TV; g.zStrideO = 128 * TV; g.oDiv = BIG; g.oStrideV = 1;
    g.M = 128; g.K = 128; g.P = 7200;
    launch_gemm(g, 32);
  }
  hipLaunchKernelGGL(sagc_apply_kernel, dim3(3600, 32), dim3(256), 0, stream,
                     bufY, bufDA, bscale(2), bshift(2), bufA, bufB);

  // ================= TAGC =================
  for (int ph = 0; ph < 2; ++ph) {
    GemmArgs g{};
    g.A = IN[23 + ph]; g.sAm = 128; g.sAk = 1;
    g.B = bufB; g.sBk = TV; g.zStrideB = 128 * TV;
    g.colDiv = BIG; g.colStrideV = 1;
    g.O = ph ? bufP2 : bufP1; g.sOm = TV; g.zStrideO = 32 * TV;
    g.oDiv = BIG; g.oStrideV = 1;
    g.M = 32; g.K = 128; g.P = 7200;
    launch_gemm(g, 32);
  }
  { // gcn conv
    GemmArgs g{};
    g.A = IN[25]; g.sAm = 128; g.sAk = 1;
    g.B = bufB; g.sBk = TV; g.zStrideB = 128 * TV;
    g.colDiv = BIG; g.colStrideV = 1;
    g.bias = IN[26];
    g.O = bufY; g.sOm = TV; g.zStrideO = 128 * TV; g.oDiv = BIG; g.oStrideV = 1;
    g.M = 128; g.K = 128; g.P = 7200;
    launch_gemm(g, 32);
  }
  for (int n = 0; n < 32; ++n) {
    { // dA[v,t,t'] = Ph^T Ps per v (z-dim = v, stride 1 element)
      GemmArgs g{};
      g.A = bufP1 + (long)n * 32 * TV; g.sAm = 25; g.sAk = TV; g.zStrideA = 1;
      g.B = bufP2 + (long)n * 32 * TV; g.sBk = TV; g.zStrideB = 1;
      g.colDiv = BIG; g.colStrideV = 25;
      g.O = bufDA; g.sOm = 288; g.zStrideO = 288 * 288; g.oDiv = BIG; g.oStrideV = 1;
      g.M = 288; g.K = 32; g.P = 288;
      launch_gemm(g, 25);
    }
    hipLaunchKernelGGL(l2norm_rows_kernel, dim3(900), dim3(256), 0, stream,
                       bufDA, 7200, 288);
    { // z[c,t,v] = Y[c,q,v] @ dA[v,q,t], fused BN + residual(x2) + relu -> x3
      GemmArgs g{};
      g.A = bufY + (long)n * 128 * TV; g.sAm = TV; g.sAk = 25; g.zStrideA = 1;
      g.B = bufDA; g.sBk = 288; g.zStrideB = 288 * 288;
      g.colDiv = BIG; g.colStrideV = 1;
      g.bnScale = bscale(3); g.bnShift = bshift(3); g.bnOut = 1;
      g.R = bufB + (long)n * 128 * TV; g.sRm = TV; g.zStrideR = 1;
      g.rDiv = BIG; g.rStrideV = 25;
      g.relu = 1;
      g.O = bufA + (long)n * 128 * TV; g.sOm = TV; g.zStrideO = 1;
      g.oDiv = BIG; g.oStrideV = 25;
      g.M = 128; g.K = 288; g.P = 288;
      launch_gemm(g, 25);
    }
  }

  // ================= MSTL blocks =================
  auto mstl_block = [&](const float* X, int Tin, int s, int pb, int s0, float* outP) {
    int Tout = (Tin - 1) / s + 1;
    long Pin = (long)Tin * 25, Pout = (long)Tout * 25;
    if (s != 1) {  // residual = BN(conv1x1(x[:,:,::s]))
      GemmArgs g{};
      g.A = IN[pb + 74]; g.sAm = 128; g.sAk = 1;
      g.B = X; g.sBk = Pin; g.zStrideB = 128 * Pin;
      g.colDiv = 25; g.colStrideT = 25 * s; g.colStrideV = 1;
      g.bias = IN[pb + 75];
      g.bnScale = bscale(s0 + 13); g.bnShift = bshift(s0 + 13); g.bnOut = 1;
      g.O = bufR; g.sOm = Pout; g.zStrideO = 128 * Pout; g.oDiv = BIG; g.oStrideV = 1;
      g.M = 128; g.K = 128; g.P = (int)Pout;
      launch_gemm(g, 32);
    }
    for (int br = 0; br < 5; ++br) {  // 4 dilated branches + maxpool branch
      int base = pb + (br < 4 ? br * 12 : 48);
      GemmArgs g{};
      g.A = IN[base]; g.sAm = 128; g.sAk = 1;
      g.B = X; g.sBk = Pin; g.zStrideB = 128 * Pin;
      g.colDiv = BIG; g.colStrideV = 1;
      g.bias = IN[base + 1];
      g.bnScale = bscale(s0 + 2 * br); g.bnShift = bshift(s0 + 2 * br);
      g.bnOut = 1; g.relu = 1;
      g.O = bufH; g.sOm = Pin; g.zStrideO = 32 * Pin; g.oDiv = BIG; g.oStrideV = 1;
      g.M = 32; g.K = 128; g.P = (int)Pin;
      launch_gemm(g, 32);
      dim3 gr((unsigned)((32 * Pout + 255) / 256), 32);
      if (br < 4)
        hipLaunchKernelGGL(tconv_kernel, gr, dim3(256), 0, stream,
                           bufH, IN[base + 6], IN[base + 7],
                           bscale(s0 + 2 * br + 1), bshift(s0 + 2 * br + 1),
                           bufC, Tin, Tout, s, br + 1, br * 32);
      else
        hipLaunchKernelGGL(maxpool_kernel, gr, dim3(256), 0, stream,
                           bufH, bscale(s0 + 9), bshift(s0 + 9),
                           bufC, Tin, Tout, s, 128);
    }
    { // branch 6: BN(conv1x1(x[:,:,::s])) -> concat[160:192]
      int base = pb + 58;
      GemmArgs g{};
      g.A = IN[base]; g.sAm = 128; g.sAk = 1;
      g.B = X; g.sBk = Pin; g.zStrideB = 128 * Pin;
      g.colDiv = 25; g.colStrideT = 25 * s; g.colStrideV = 1;
      g.bias = IN[base + 1];
      g.bnScale = bscale(s0 + 10); g.bnShift = bshift(s0 + 10); g.bnOut = 1;
      g.O = bufC + 160 * Pout; g.sOm = Pout; g.zStrideO = 192 * Pout;
      g.oDiv = BIG; g.oStrideV = 1;
      g.M = 32; g.K = 128; g.P = (int)Pout;
      launch_gemm(g, 32);
    }
    { // fuse conv: relu(BN_in(concat)) folded into B-load; +bias, BN_out, +res
      GemmArgs g{};
      g.A = IN[pb + 68]; g.sAm = 192; g.sAk = 1;
      g.B = bufC; g.sBk = Pout; g.zStrideB = 192 * Pout;
      g.colDiv = BIG; g.colStrideV = 1;
      g.inFold = 1; g.inScale = bscale(s0 + 11); g.inShift = bshift(s0 + 11);
      g.bias = IN[pb + 69];
      g.bnScale = bscale(s0 + 12); g.bnShift = bshift(s0 + 12); g.bnOut = 1;
      g.R = (s != 1) ? bufR : X;
      g.sRm = Pout; g.zStrideR = 128 * Pout; g.rDiv = BIG; g.rStrideV = 1;
      g.O = outP; g.sOm = Pout; g.zStrideO = 128 * Pout; g.oDiv = BIG; g.oStrideV = 1;
      g.M = 128; g.K = 192; g.P = (int)Pout;
      launch_gemm(g, 32);
    }
  };

  mstl_block(bufA, 288, 2, 31, 4, bufB);             // MSTL1 (stride 2) -> bufB
  mstl_block(bufB, 144, 1, 111, 18, (float*)d_out);  // MSTL2 (stride 1) -> d_out
  (void)in_sizes; (void)out_size; (void)ws_size;
}
